// GraphColorizer_33182917329408
// MI455X (gfx1250) — matrix operand order repaired and checked by reading, not compiled, not run on hardware
//
#include <hip/hip_runtime.h>
#include <hip/hip_bf16.h>
#include <math.h>

typedef __attribute__((ext_vector_type(8)))  float  v8f;
typedef __attribute__((ext_vector_type(8)))  __bf16 v8bf;
typedef __attribute__((ext_vector_type(16))) __bf16 v16bf;

#define N_VERT 8192
#define EMB    512
#define H1DIM  512
#define H2DIM  400
#define H2PAD  416   // 400 padded to multiple of 32 (K of GEMM3)
#define NCLS   257
#define DEG    32

__device__ __forceinline__ unsigned short f2bf(float f) {
  unsigned u = __float_as_uint(f);
  unsigned r = u + 0x7FFFu + ((u >> 16) & 1u);   // round-to-nearest-even
  return (unsigned short)(r >> 16);
}

// ---- CDNA5 async copy: global -> LDS, 16B per lane, tracked by ASYNCcnt ----
__device__ __forceinline__ void async_copy_b128(unsigned lds_off, const void* gaddr) {
  asm volatile("global_load_async_to_lds_b128 %0, %1, off"
               :: "v"(lds_off), "v"(gaddr) : "memory");
}
template<int N>
__device__ __forceinline__ void wait_asynccnt() {
#if __has_builtin(__builtin_amdgcn_s_wait_asynccnt)
  __builtin_amdgcn_s_wait_asynccnt(N);
#else
  asm volatile("s_wait_asynccnt %0" :: "i"(N) : "memory");
#endif
}

// ---- graph_emb = mean(embeddings, axis=0), two-stage reduction ----
__global__ __launch_bounds__(256) void mean_partial(const float* __restrict__ emb,
                                                    float* __restrict__ part) {
  const int t = threadIdx.x, b = blockIdx.x;
  float s0 = 0.f, s1 = 0.f;
  const float* base = emb + (size_t)b * 256 * EMB;
  for (int r = 0; r < 256; ++r) {
    s0 += base[(size_t)r * EMB + t];
    s1 += base[(size_t)r * EMB + t + 256];
  }
  part[b * EMB + t]       = s0;
  part[b * EMB + t + 256] = s1;
}

__global__ __launch_bounds__(256) void mean_final(const float* __restrict__ part,
                                                  float* __restrict__ ge) {
  const int j = blockIdx.x * 256 + threadIdx.x;
  float s = 0.f;
  for (int b = 0; b < 32; ++b) s += part[b * EMB + j];
  ge[j] = s * (1.0f / 8192.0f);
}

// ---- fold constant half of layer1: cv = b1 + graph_emb @ W1[512:,:] ----
__global__ __launch_bounds__(256) void cvec_kernel(const float* __restrict__ ge,
                                                   const float* __restrict__ W1,
                                                   const float* __restrict__ b1,
                                                   float* __restrict__ cv) {
  const int j = blockIdx.x * 256 + threadIdx.x;
  float c = b1[j];
  for (int i = 0; i < EMB; ++i)
    c += ge[i] * W1[(size_t)(EMB + i) * H1DIM + j];
  cv[j] = c;
}

// ---- elementwise fp32 -> bf16 (embeddings) ----
__global__ __launch_bounds__(256) void conv_bf16(const float* __restrict__ src,
                                                 unsigned short* __restrict__ dst,
                                                 int n4) {
  int i = blockIdx.x * 256 + threadIdx.x;
  if (i < n4) {
    float4 v = *(const float4*)(src + (size_t)i * 4);
    uint2 pk;
    pk.x = (unsigned)f2bf(v.x) | ((unsigned)f2bf(v.y) << 16);
    pk.y = (unsigned)f2bf(v.z) | ((unsigned)f2bf(v.w) << 16);
    *(uint2*)(dst + (size_t)i * 4) = pk;
  }
}

// ---- Bt[n][k] = B[k][n] (fp32 -> bf16), zero padding to NP x KP ----
__global__ __launch_bounds__(256) void transpose_bf16(const float* __restrict__ B, int ldb,
                                                      int N, int K,
                                                      unsigned short* __restrict__ Bt, int KP) {
  __shared__ float t[32][33];
  const int nb = blockIdx.x * 32, kb = blockIdx.y * 32;
  const int tx = threadIdx.x & 31, ty = threadIdx.x >> 5;   // 32 x 8
  #pragma unroll
  for (int r = 0; r < 32; r += 8) {
    int k = kb + ty + r, n = nb + tx;
    t[ty + r][tx] = (k < K && n < N) ? B[(size_t)k * ldb + n] : 0.f;
  }
  __syncthreads();
  #pragma unroll
  for (int r = 0; r < 32; r += 8) {
    int n = nb + ty + r, k = kb + tx;
    Bt[(size_t)n * KP + k] = f2bf(t[tx][ty + r]);
  }
}

// ---- bf16 WMMA GEMM, async-staged + double-buffered LDS ----
// out[M,NStore] = act(A[M,K]bf16 @ Bt[N,K]^T bf16 + bias[N]); all dims compile-time.
// 256 threads = 8 waves; block tile 128x64; wave tile 32x32 (2x2 of 16x16x32).
template<int ACT, int OUTBF, int NREAL, int NSTORE, int K, int LDA, int LDB, int LDO>
__global__ __launch_bounds__(256) void gemm_bf16_async(
    const unsigned short* __restrict__ A,
    const unsigned short* __restrict__ Bt,
    const float* __restrict__ bias,
    void* __restrict__ outv)
{
  __shared__ unsigned short Alds[2][128 * 48];  // stride 48 bf16 (96B, 16B-aligned frags)
  __shared__ unsigned short Blds[2][64 * 48];   // (n,k) layout: contiguous K runs

  const int tid  = threadIdx.x;
  const int lane = tid & 31;
  const int wid  = tid >> 5;
  const int half = lane >> 4;
  const int l16  = lane & 15;
  const int wm   = (wid & 3) * 32;
  const int wn   = (wid >> 2) * 32;
  const int m0   = blockIdx.y * 128;
  const int n0   = blockIdx.x * 64;

  const unsigned aBase = (unsigned)(size_t)&Alds[0][0];
  const unsigned bBase = (unsigned)(size_t)&Blds[0][0];
  const int cr = tid >> 2;            // chunk row 0..63
  const int ck = (tid & 3) * 8;       // k offset of 8-bf16 (16B) chunk

  auto issue = [&](int k0, int buf) {
    const unsigned aO = aBase + (unsigned)buf * (128 * 48 * 2);
    const unsigned bO = bBase + (unsigned)buf * (64 * 48 * 2);
    async_copy_b128(aO + (unsigned)((cr)      * 48 + ck) * 2, A  + (size_t)(m0 + cr)      * LDA + k0 + ck);
    async_copy_b128(aO + (unsigned)((cr + 64) * 48 + ck) * 2, A  + (size_t)(m0 + cr + 64) * LDA + k0 + ck);
    async_copy_b128(bO + (unsigned)((cr)      * 48 + ck) * 2, Bt + (size_t)(n0 + cr)      * LDB + k0 + ck);
  };

  v8f acc[2][2] = {};
  constexpr int S = K / 32;
  issue(0, 0);
  #pragma unroll
  for (int s = 0; s < S; ++s) {
    const int buf = s & 1;
    if (s + 1 < S) { issue((s + 1) * 32, buf ^ 1); wait_asynccnt<3>(); }
    else           { wait_asynccnt<0>(); }
    __syncthreads();                     // stage s visible to all waves
    const unsigned short* Ab = &Alds[buf][0];
    const unsigned short* Bb = &Blds[buf][0];
    #pragma unroll
    for (int tm = 0; tm < 2; ++tm) {
      // A 16x32 bf16 frag: elems 0-15: K=half*16+0..15
      int ar = (wm + tm * 16 + l16) * 48 + half * 16;
      union { v16bf v; v8bf h[2]; } a;
      a.h[0] = *(const v8bf*)&Ab[ar];
      a.h[1] = *(const v8bf*)&Ab[ar + 8];
      #pragma unroll
      for (int tn = 0; tn < 2; ++tn) {
        // B 32x16 frag: lane = col n, contiguous K = half*16..+15
        const v16bf b = *(const v16bf*)&Bb[(wn + tn * 16 + l16) * 48 + half * 16];
        acc[tm][tn] = __builtin_amdgcn_wmma_f32_16x16x32_bf16(
            false, a.v, false, b, (short)0, acc[tm][tn], false, false);
      }
    }
    __syncthreads();                     // all reads of buf done before reuse
  }

  // Epilogue: C VGPR j -> row = half*8 + j, col = l16
  #pragma unroll
  for (int tm = 0; tm < 2; ++tm) {
    #pragma unroll
    for (int tn = 0; tn < 2; ++tn) {
      int n = n0 + wn + tn * 16 + l16;
      if (n < NSTORE) {
        float bv = (n < NREAL) ? bias[n] : 0.f;
        #pragma unroll
        for (int j = 0; j < 8; ++j) {
          int m = m0 + wm + tm * 16 + half * 8 + j;
          float v = acc[tm][tn][j] + bv;
          if (ACT) v = (v >= 0.f) ? v : 0.01f * v;
          if (n >= NREAL) v = 0.f;                    // deterministic zero padding
          if (OUTBF) ((unsigned short*)outv)[(size_t)m * LDO + n] = f2bf(v);
          else       ((float*)outv)[(size_t)m * LDO + n] = v;
        }
      }
    }
  }
}

// ---- sequential coloring: one wave32, colors resident in LDS ----
__device__ __forceinline__ float rng01(unsigned x) {
  x ^= x >> 17; x *= 0xed5ad4bbu;
  x ^= x >> 11; x *= 0xac4c1b51u;
  x ^= x >> 15; x *= 0x31848babu;
  x ^= x >> 14;
  return ((float)(x >> 8) + 0.5f) * (1.0f / 16777216.0f);
}

__global__ __launch_bounds__(32) void seq_color(
    const float* __restrict__ logits,   // [N_VERT, 257]
    const int*   __restrict__ nbr,      // [N_VERT, 32]
    const float* __restrict__ baseline, // [1]
    float*       __restrict__ out)      // [8193]
{
  __shared__ int colors[N_VERT];
  __shared__ unsigned maskw[9];
  const int lane = threadIdx.x;
  const float NEG = -1e30f;

  for (int i = lane; i < N_VERT; i += 32) colors[i] = -1;
  __syncthreads();
  if (lane == 0) colors[0] = 0;
  __syncthreads();

  int n_used = 1;
  float logp = 0.f;

  for (int v = 1; v < N_VERT; ++v) {
    if (v + 1 < N_VERT && lane < 9)     // hide next row's fetch (global_prefetch_b8)
      __builtin_prefetch((const char*)&logits[(size_t)(v + 1) * NCLS] + lane * 128, 0, 0);
    if (lane < 9) maskw[lane] = 0u;
    __syncthreads();
    int nb = nbr[v * DEG + lane];
    int nc = colors[nb];
    if (nc >= 0) atomicOr(&maskw[nc >> 5], 1u << (nc & 31));
    __syncthreads();

    float vals[9];
    float mx = NEG, best = NEG;
    int bidx = 0;
    #pragma unroll
    for (int s = 0; s < 9; ++s) {
      int col = lane + s * 32;
      float lg = NEG;
      if (col < NCLS) {
        lg = logits[(size_t)v * NCLS + col];
        bool irrel = (col >= n_used) && (col < 256);
        bool adj = (maskw[col >> 5] >> (col & 31)) & 1u;
        if (irrel || adj) lg = NEG;
      }
      vals[s] = lg;
      if (lg > mx) mx = lg;
      if (lg > -1e29f) {
        float g = -__logf(-__logf(rng01((unsigned)(v * 521 + col) * 2654435761u + 12345u)));
        float sc = lg + g;
        if (sc > best) { best = sc; bidx = col; }
      }
    }
    #pragma unroll
    for (int off = 16; off > 0; off >>= 1) {
      float ob = __shfl_xor(best, off, 32);
      int   oi = __shfl_xor(bidx, off, 32);
      if (ob > best || (ob == best && oi < bidx)) { best = ob; bidx = oi; }
      float om = __shfl_xor(mx, off, 32);
      if (om > mx) mx = om;
    }
    float se = 0.f, lraw = NEG;
    #pragma unroll
    for (int s = 0; s < 9; ++s) {
      se += __expf(vals[s] - mx);
      if (lane + s * 32 == bidx) lraw = vals[s];
    }
    #pragma unroll
    for (int off = 16; off > 0; off >>= 1) {
      se += __shfl_xor(se, off, 32);
      float ol = __shfl_xor(lraw, off, 32);
      if (ol > lraw) lraw = ol;
    }
    float p = __expf(lraw - mx) / se;
    logp += __logf(p + 1e-8f) - __logf(1e-8f);

    int raw = bidx;
    int chosen = (raw == 256) ? n_used : raw;
    if (raw == 256) n_used++;
    if (lane == 0) colors[v] = chosen;
    __syncthreads();
  }

  for (int i = lane; i < N_VERT; i += 32) out[i] = (float)colors[i];
  if (lane == 0)
    out[N_VERT] = ((float)n_used - baseline[0]) * logp * (1.0f / 8192.0f);
}

extern "C" void kernel_launch(void* const* d_in, const int* in_sizes, int n_in,
                              void* d_out, int out_size, void* d_ws, size_t ws_size,
                              hipStream_t stream) {
  (void)in_sizes; (void)n_in; (void)out_size; (void)ws_size;
  const float* emb = (const float*)d_in[0];
  const int*   nbr = (const int*)d_in[1];
  const float* W1  = (const float*)d_in[2];
  const float* b1  = (const float*)d_in[3];
  const float* W2  = (const float*)d_in[4];
  const float* b2  = (const float*)d_in[5];
  const float* W3  = (const float*)d_in[6];
  const float* b3  = (const float*)d_in[7];
  const float* baseline = (const float*)d_in[8];

  char* ws = (char*)d_ws;
  float*          GE   = (float*)(ws + 0);                        // 512 f32
  float*          PT   = (float*)(ws + 2048);                     // 32*512 f32
  float*          CV   = (float*)(ws + 67584);                    // 512 f32
  unsigned short* EMBb = (unsigned short*)(ws + 69632);           // 8192x512 bf16
  unsigned short* Bt1  = (unsigned short*)(ws + 8458240);         // 512x512 bf16
  unsigned short* Bt2  = (unsigned short*)(ws + 8982528);         // 448x512 bf16
  unsigned short* Bt3  = (unsigned short*)(ws + 9441280);         // 320x416 bf16
  unsigned short* H1b  = (unsigned short*)(ws + 9707520);         // 8192x512 bf16
  unsigned short* H2b  = (unsigned short*)(ws + 18096128);        // 8192x416 bf16 (padded)
  float*          LG   = (float*)(ws + 24911872);                 // 8192x257 f32
  float* outf = (float*)d_out;

  mean_partial<<<32, 256, 0, stream>>>(emb, PT);
  mean_final<<<2, 256, 0, stream>>>(PT, GE);
  cvec_kernel<<<2, 256, 0, stream>>>(GE, W1, b1, CV);

  // One-time precision/layout conversion (bf16 + transposed/zero-padded weights)
  conv_bf16<<<4096, 256, 0, stream>>>(emb, EMBb, (N_VERT * EMB) / 4);
  transpose_bf16<<<dim3(16, 16), 256, 0, stream>>>(W1, H1DIM, H1DIM, EMB,   Bt1, 512);
  transpose_bf16<<<dim3(14, 16), 256, 0, stream>>>(W2, H2DIM, H2DIM, H1DIM, Bt2, 512);
  transpose_bf16<<<dim3(10, 13), 256, 0, stream>>>(W3, NCLS,  NCLS,  H2DIM, Bt3, H2PAD);

  // logits pipeline (all-async staged WMMA GEMMs)
  gemm_bf16_async<1, 1, 512, 512, 512, 512, 512, 512>
      <<<dim3(8, 64), 256, 0, stream>>>(EMBb, Bt1, CV, H1b);
  gemm_bf16_async<1, 1, 400, 416, 512, 512, 512, 416>
      <<<dim3(7, 64), 256, 0, stream>>>(H1b, Bt2, b2, H2b);
  gemm_bf16_async<0, 0, 257, 257, 416, 416, 416, 257>
      <<<dim3(5, 64), 256, 0, stream>>>(H2b, Bt3, b3, LG);

  seq_color<<<1, 32, 0, stream>>>(LG, nbr, baseline, outf);
}
